// CausalSelfAttention_26396869001390
// MI455X (gfx1250) — matrix operand order repaired and checked by reading, not compiled, not run on hardware
//
#include <hip/hip_runtime.h>
#include <hip/hip_bf16.h>

typedef __attribute__((ext_vector_type(16))) _Float16 v16h;
typedef __attribute__((ext_vector_type(8)))  _Float16 v8h;
typedef __attribute__((ext_vector_type(8)))  float    v8f;
typedef __attribute__((ext_vector_type(4)))  int      v4i;

#define BATCH 4
#define SEQ   2048
#define DMODEL 512
#define NHEAD 8
#define DHEAD 64
#define MTOT  (BATCH*SEQ)          // 8192

#ifndef __has_builtin
#define __has_builtin(x) 0
#endif

#if __has_builtin(__builtin_amdgcn_global_load_async_to_lds_b128)
#define HAVE_ASYNC_LDS 1
#else
#define HAVE_ASYNC_LDS 0
#endif

#if HAVE_ASYNC_LDS
typedef v4i __attribute__((address_space(1)))* gv4i_p;   // global <4 x i32>*
typedef v4i __attribute__((address_space(3)))* lv4i_p;   // LDS    <4 x i32>*
#if __has_builtin(__builtin_amdgcn_s_wait_asynccnt)
#define ASYNC_WAIT(n) __builtin_amdgcn_s_wait_asynccnt(n)
#else
#define ASYNC_WAIT(n) asm volatile("s_wait_asynccnt %0" :: "i"(n) : "memory")
#endif
#else
#define ASYNC_WAIT(n) do {} while (0)
#endif

// 16-byte global -> LDS copy: async (gfx1250 GLOBAL_LOAD_ASYNC_TO_LDS_B128,
// tracked by ASYNCcnt, no VGPR round trip) when available, else sync fallback.
__device__ __forceinline__ void ld_async_b128(void* lds, const void* gmem) {
#if HAVE_ASYNC_LDS
    void* g = const_cast<void*>(gmem);
    __builtin_amdgcn_global_load_async_to_lds_b128((gv4i_p)g, (lv4i_p)lds, 0, 0);
#else
    *(v8h*)lds = *(const v8h*)gmem;
#endif
}

// ---------------------------------------------------------------------------
// f32 -> f16 conversion (grid-stride)
// ---------------------------------------------------------------------------
__global__ __launch_bounds__(256) void cvt_f32_f16(const float* __restrict__ in,
                                                   _Float16* __restrict__ out, int n) {
    int i = blockIdx.x * blockDim.x + threadIdx.x;
    int stride = gridDim.x * blockDim.x;
    for (; i < n; i += stride) out[i] = (_Float16)in[i];
}

// ---------------------------------------------------------------------------
// GEMM  C = A(MxK) @ W(NxK)^T   with WMMA f16->f32, double-buffered async
// LDS staging (prefetch chunk i+1 while computing chunk i).
// mode 0: scatter f16 results into Q/K/V (B,NH,T,DH); Q scaled by qscale.
// mode 1: f32 output = acc + bias[n], row-major MxN.
// Block tile: 64(M) x 128(N); 256 threads = 8 waves (4x2), wave tile 16x64.
// ---------------------------------------------------------------------------
__global__ __launch_bounds__(256)
void gemm_xwT(const _Float16* __restrict__ A, const _Float16* __restrict__ W,
              int Nsize, int Ksize, int mode,
              _Float16* __restrict__ qout, _Float16* __restrict__ kout,
              _Float16* __restrict__ vout,
              const float* __restrict__ bias, float* __restrict__ out,
              float qscale) {
    __shared__ __align__(16) _Float16 Asm[2][64 * 32];
    __shared__ __align__(16) _Float16 Wsm[2][128 * 32];

    const int tid  = threadIdx.x;
    const int lane = tid & 31;
    const int wave = tid >> 5;            // 0..7
    const int wm   = wave & 3;            // M sub-tile
    const int wn   = wave >> 2;           // N sub-tile (0/1)
    const int g    = (lane >> 4) & 1;
    const int nl   = lane & 15;

    const int nblk = Nsize / 128;
    const int m0   = (blockIdx.x / nblk) * 64;
    const int n0   = (blockIdx.x % nblk) * 128;

    // per-thread staging coordinates (3 async b128 per thread per chunk)
    const int ar = tid >> 2, ac = (tid & 3) * 8;
    const int wr = tid >> 1, wc = (tid & 1) * 16;

    auto stage = [&](int buf, int k0) {
        ld_async_b128(&Asm[buf][ar * 32 + ac],
                      A + (size_t)(m0 + ar) * Ksize + k0 + ac);
        const _Float16* src = W + (size_t)(n0 + wr) * Ksize + k0 + wc;
        ld_async_b128(&Wsm[buf][wr * 32 + wc], src);
        ld_async_b128(&Wsm[buf][wr * 32 + wc + 8], src + 8);
    };

    v8f acc[4] = {};
    const int nch = Ksize / 32;
    stage(0, 0);

    for (int i = 0; i < nch; ++i) {
        const int cur = i & 1;
        if (i + 1 < nch) {
            stage(cur ^ 1, 32 * (i + 1));   // prefetch next chunk
            ASYNC_WAIT(3);                  // wait only for current chunk
        } else {
            ASYNC_WAIT(0);
        }
        __syncthreads();                    // current chunk visible to all waves

        // A fragment (16x32): lane = M row, element h -> K
        v16h af;
        {
            const _Float16* ab = &Asm[cur][(16 * wm + nl) * 32];
            #pragma unroll
            for (int h = 0; h < 16; ++h) {
                int k = 16 * g + h;
                af[h] = ab[k];
            }
        }
        #pragma unroll
        for (int nt = 0; nt < 4; ++nt) {
            // B fragment from W (NxK): element h -> K = 16*g + h, contiguous
            const _Float16* wb = &Wsm[cur][(64 * wn + 16 * nt + nl) * 32 + 16 * g];
            v16h bf;
            #pragma unroll
            for (int h = 0; h < 16; ++h) bf[h] = wb[h];
            acc[nt] = __builtin_amdgcn_wmma_f32_16x16x32_f16(
                false, af, false, bf, (short)0, acc[nt], false, false);
        }
        __syncthreads();                    // reads done before buffer reuse
    }

    if (mode == 0) {
        #pragma unroll
        for (int nt = 0; nt < 4; ++nt) {
            int n     = n0 + 64 * wn + 16 * nt + nl;
            int which = n / DMODEL;                 // 0=q 1=k 2=v (uniform per tile)
            int d     = n % DMODEL;
            int h     = d / DHEAD, dh = d % DHEAD;
            _Float16* base = (which == 0) ? qout : ((which == 1) ? kout : vout);
            float sc = (which == 0) ? qscale : 1.0f;
            #pragma unroll
            for (int r = 0; r < 8; ++r) {
                int m = m0 + 16 * wm + r + 8 * g;
                int b = m >> 11, t = m & (SEQ - 1);
                base[(((size_t)b * NHEAD + h) * SEQ + t) * DHEAD + dh] =
                    (_Float16)(acc[nt][r] * sc);
            }
        }
    } else {
        #pragma unroll
        for (int nt = 0; nt < 4; ++nt) {
            int n  = n0 + 64 * wn + 16 * nt + nl;
            float bv = bias[n];
            #pragma unroll
            for (int r = 0; r < 8; ++r) {
                int m = m0 + 16 * wm + r + 8 * g;
                out[(size_t)m * Nsize + n] = acc[nt][r] + bv;
            }
        }
    }
}

// ---------------------------------------------------------------------------
// Flash attention: grid (T/64, B*NH), 128 threads = 4 waves.
// Wave w owns q rows [q0+16w, q0+16w+15], full DH=64 output (4 accum tiles).
// Streams keys/values in 32-chunks (double-buffered async LDS staging) with
// online softmax. Q pre-scaled by 1/sqrt(DH).
// ---------------------------------------------------------------------------
__global__ __launch_bounds__(128)
void attn_fwd(const _Float16* __restrict__ Q, const _Float16* __restrict__ K,
              const _Float16* __restrict__ V, _Float16* __restrict__ Y) {
    __shared__ __align__(16) _Float16 Qs[64 * 64];
    __shared__ __align__(16) _Float16 Ks[2][32 * 64];
    __shared__ __align__(16) _Float16 Vs[2][32 * 64];
    __shared__ __align__(16) _Float16 Ps[4][16 * 32];

    const int tid  = threadIdx.x;
    const int lane = tid & 31;
    const int wave = tid >> 5;              // 0..3
    const int g    = (lane >> 4) & 1;
    const int nl   = lane & 15;
    const int bh   = blockIdx.y;            // b*NH + h
    const int q0   = blockIdx.x * 64;

    const _Float16* Qh = Q + (size_t)bh * SEQ * DHEAD;
    const _Float16* Kh = K + (size_t)bh * SEQ * DHEAD;
    const _Float16* Vh = V + (size_t)bh * SEQ * DHEAD;

    // 4 async b128 per thread per chunk (K and V tiles, 32x64 each)
    auto stageKV = [&](int buf, int k0) {
        for (int i = tid; i < (32 * 64) / 8; i += 128) {
            ld_async_b128(&((v8h*)Ks[buf])[i],
                          &((const v8h*)(Kh + (size_t)k0 * DHEAD))[i]);
            ld_async_b128(&((v8h*)Vs[buf])[i],
                          &((const v8h*)(Vh + (size_t)k0 * DHEAD))[i]);
        }
    };

    // stage Q tile 64x64 (row-major, ld=64) + first K/V chunk
    for (int i = tid; i < (64 * 64) / 8; i += 128)
        ld_async_b128(&((v8h*)Qs)[i], &((const v8h*)(Qh + (size_t)q0 * DHEAD))[i]);
    stageKV(0, 0);
    ASYNC_WAIT(4);                          // Q landed (first 4 loads/thread)
    __syncthreads();

    // Q fragments (2 K-chunks of 32) held in registers for whole loop
    v16h qf[2];
    {
        const _Float16* qb = &Qs[(16 * wave + nl) * 64];
        #pragma unroll
        for (int kc = 0; kc < 2; ++kc)
            #pragma unroll
            for (int h = 0; h < 16; ++h) {
                int k = 32 * kc + 16 * g + h;
                qf[kc][h] = qb[k];
            }
    }

    v8f o[4] = {};
    float mrow[8], lrow[8];
    #pragma unroll
    for (int r = 0; r < 8; ++r) { mrow[r] = -1e30f; lrow[r] = 0.0f; }

    const int nit = (q0 + 64) / 32;         // causal: keys [0, q0+64)
    for (int it = 0; it < nit; ++it) {
        const int cur = it & 1;
        const int k0  = 32 * it;
        if (it + 1 < nit) {
            stageKV(cur ^ 1, 32 * (it + 1));  // prefetch next K/V chunk
            ASYNC_WAIT(4);                    // wait only for current chunk
        } else {
            ASYNC_WAIT(0);
        }
        __syncthreads();

        // S(16x32) = Q(16x64) @ K(32x64)^T  -- 2 n-tiles x 2 k-chunks
        v8f s[2] = {};
        #pragma unroll
        for (int nt = 0; nt < 2; ++nt)
            #pragma unroll
            for (int kc = 0; kc < 2; ++kc) {
                const _Float16* kb = &Ks[cur][(16 * nt + nl) * 64 + 32 * kc + 16 * g];
                v16h bf;
                #pragma unroll
                for (int h = 0; h < 16; ++h) bf[h] = kb[h];
                s[nt] = __builtin_amdgcn_wmma_f32_16x16x32_f16(
                    false, qf[kc], false, bf, (short)0, s[nt], false, false);
            }

        // causal mask + online softmax (row = r + 8g, col = lane&15)
        #pragma unroll
        for (int r = 0; r < 8; ++r) {
            int q = q0 + 16 * wave + r + 8 * g;
            float v0 = (k0 + nl      <= q) ? s[0][r] : -1e30f;
            float v1 = (k0 + 16 + nl <= q) ? s[1][r] : -1e30f;
            float mx = fmaxf(v0, v1);
            mx = fmaxf(mx, __shfl_xor(mx, 1, 16));
            mx = fmaxf(mx, __shfl_xor(mx, 2, 16));
            mx = fmaxf(mx, __shfl_xor(mx, 4, 16));
            mx = fmaxf(mx, __shfl_xor(mx, 8, 16));
            float mnew = fmaxf(mrow[r], mx);
            float corr = __expf(mrow[r] - mnew);
            float p0 = __expf(v0 - mnew);
            float p1 = __expf(v1 - mnew);
            float ls = p0 + p1;
            ls += __shfl_xor(ls, 1, 16);
            ls += __shfl_xor(ls, 2, 16);
            ls += __shfl_xor(ls, 4, 16);
            ls += __shfl_xor(ls, 8, 16);
            lrow[r] = lrow[r] * corr + ls;
            mrow[r] = mnew;
            #pragma unroll
            for (int nt = 0; nt < 4; ++nt) o[nt][r] *= corr;
            // accumulator layout -> row-major P in per-wave LDS (16x32, ld=32)
            Ps[wave][(r + 8 * g) * 32 + nl]      = (_Float16)p0;
            Ps[wave][(r + 8 * g) * 32 + 16 + nl] = (_Float16)p1;
        }

        // O(16x64) += P(16x32) @ V(32x64)
        v16h pf;
        {
            const _Float16* pb = &Ps[wave][nl * 32];
            #pragma unroll
            for (int h = 0; h < 16; ++h) {
                int k = 16 * g + h;
                pf[h] = pb[k];
            }
        }
        #pragma unroll
        for (int nt = 0; nt < 4; ++nt) {
            v16h bf;
            #pragma unroll
            for (int h = 0; h < 16; ++h)
                bf[h] = Vs[cur][(h + 16 * g) * 64 + 16 * nt + nl];
            o[nt] = __builtin_amdgcn_wmma_f32_16x16x32_f16(
                false, pf, false, bf, (short)0, o[nt], false, false);
        }
        __syncthreads();                    // reads done before buffer reuse
    }

    // epilogue: normalize and store y as (B, T, NH*DH) row-major (== (B,T,D))
    const int b = bh >> 3, h = bh & 7;
    #pragma unroll
    for (int nt = 0; nt < 4; ++nt)
        #pragma unroll
        for (int r = 0; r < 8; ++r) {
            int q  = q0 + 16 * wave + r + 8 * g;
            int dh = 16 * nt + nl;
            float val = o[nt][r] / lrow[r];
            Y[(((size_t)b * SEQ + q) * NHEAD + h) * DHEAD + dh] = (_Float16)val;
        }
}

// ---------------------------------------------------------------------------
extern "C" void kernel_launch(void* const* d_in, const int* in_sizes, int n_in,
                              void* d_out, int out_size, void* d_ws, size_t ws_size,
                              hipStream_t stream) {
    const float* x     = (const float*)d_in[0];
    const float* Wqkv  = (const float*)d_in[1];
    const float* Wout  = (const float*)d_in[2];
    const float* bout  = (const float*)d_in[3];
    float*       out   = (float*)d_out;

    _Float16* ws    = (_Float16*)d_ws;
    _Float16* xh    = ws;                                  // 8192*512
    _Float16* wqkvh = xh    + (size_t)MTOT * DMODEL;       // 1536*512
    _Float16* wouth = wqkvh + (size_t)3 * DMODEL * DMODEL; // 512*512
    _Float16* qh    = wouth + (size_t)DMODEL * DMODEL;     // 8192*512 each
    _Float16* kh    = qh    + (size_t)MTOT * DMODEL;
    _Float16* vh    = kh    + (size_t)MTOT * DMODEL;
    _Float16* yh    = vh    + (size_t)MTOT * DMODEL;

    cvt_f32_f16<<<2048, 256, 0, stream>>>(x, xh, MTOT * DMODEL);
    cvt_f32_f16<<<1024, 256, 0, stream>>>(Wqkv, wqkvh, 3 * DMODEL * DMODEL);
    cvt_f32_f16<<<512, 256, 0, stream>>>(Wout, wouth, DMODEL * DMODEL);

    // QKV projection: M=8192, N=1536, K=512 -> scatter Q/K/V, scale Q by 1/8
    gemm_xwT<<<(MTOT / 64) * (3 * DMODEL / 128), 256, 0, stream>>>(
        xh, wqkvh, 3 * DMODEL, DMODEL, 0, qh, kh, vh, nullptr, nullptr, 0.125f);

    // Flash attention: (T/64) x (B*NH)
    attn_fwd<<<dim3(SEQ / 64, BATCH * NHEAD), 128, 0, stream>>>(qh, kh, vh, yh);

    // Output projection: M=8192, N=512, K=512, +bias, f32 out
    gemm_xwT<<<(MTOT / 64) * (DMODEL / 128), 256, 0, stream>>>(
        yh, wouth, DMODEL, DMODEL, 1, nullptr, nullptr, nullptr, bout, out, 1.0f);
}
